// CSLoss_16423954940464
// MI455X (gfx1250) — matrix operand order repaired and checked by reading, not compiled, not run on hardware
//
#include <hip/hip_runtime.h>

// ---------------------------------------------------------------------------
// Contrastive (InfoNCE) loss, fused:  normalize -> f16 hi/lo split ->
// WMMA GEMM with in-register log-sum-exp (fixed max bound) -> mean loss.
// Target: gfx1250 (MI455X), wave32, v_wmma_f32_16x16x32_f16.
// Round 2: 2 N-tiles per wave (A-fragment reuse, higher WMMA:VMEM ratio).
// ---------------------------------------------------------------------------

#define NROWS 8192
#define DIMK  768
#define NTILES (NROWS / 16)            // 512 column tiles of 16
#define NSLICES 8                      // column slices per row-group
#define TILES_PER_SLICE (NTILES / NSLICES)  // 64 (even -> pairs of tiles)

typedef __attribute__((ext_vector_type(16))) _Float16 v16h;
typedef __attribute__((ext_vector_type(8)))  _Float16 v8h;
typedef __attribute__((ext_vector_type(8)))  float    v8f;

static constexpr float INV_TEMP     = 1.0f / 0.07f;     // 14.2857...
static constexpr float RMAX         = 1.0f / 0.07f;     // |logit| <= 1/temp (cos in [-1,1])
static constexpr float LO_SCALE     = 2048.0f;          // keep residuals out of f16 denormals
static constexpr float INV_LO_SCALE = 1.0f / 2048.0f;

// ---------------------------------------------------------------------------
// Kernel 1: row L2-normalize (eps clamp like the reference), split each value
// into f16 hi + f16 (residual * 2048). Also zero-inits the per-row S/D arrays
// so every graph replay starts clean.
// One 256-thread block per row (768 = 3 * 256). blockIdx < 8192 -> f1, else f2.
// ---------------------------------------------------------------------------
__global__ __launch_bounds__(256) void cs_normalize_split(
    const float* __restrict__ f1, const float* __restrict__ f2,
    _Float16* __restrict__ Ah, _Float16* __restrict__ Al,
    _Float16* __restrict__ Bh, _Float16* __restrict__ Bl,
    float* __restrict__ Srow, float* __restrict__ Drow)
{
    __shared__ float red[8];
    __shared__ float s_inv;

    const int b    = blockIdx.x;
    const bool is2 = (b >= NROWS);
    const int r    = is2 ? (b - NROWS) : b;
    const float* __restrict__ src = (is2 ? f2 : f1) + (size_t)r * DIMK;
    _Float16* __restrict__ dh = (is2 ? Bh : Ah) + (size_t)r * DIMK;
    _Float16* __restrict__ dl = (is2 ? Bl : Al) + (size_t)r * DIMK;

    const int tid = threadIdx.x;
    float x0 = src[tid];
    float x1 = src[tid + 256];
    float x2 = src[tid + 512];
    float ss = x0 * x0 + x1 * x1 + x2 * x2;

    #pragma unroll
    for (int mk = 1; mk < 32; mk <<= 1) ss += __shfl_xor(ss, mk, 32);
    if ((tid & 31) == 0) red[tid >> 5] = ss;
    __syncthreads();
    if (tid == 0) {
        float t = 0.0f;
        #pragma unroll
        for (int i = 0; i < 8; ++i) t += red[i];
        s_inv = 1.0f / fmaxf(sqrtf(t), 1e-8f);
        if (!is2) { Srow[r] = 0.0f; Drow[r] = 0.0f; }
    }
    __syncthreads();
    const float inv = s_inv;

    float xs[3] = {x0, x1, x2};
    #pragma unroll
    for (int i = 0; i < 3; ++i) {
        float y = xs[i] * inv;
        _Float16 hi = (_Float16)y;
        _Float16 lo = (_Float16)((y - (float)hi) * LO_SCALE);
        dh[tid + 256 * i] = hi;
        dl[tid + 256 * i] = lo;
    }
}

// ---------------------------------------------------------------------------
// Kernel 2: fused GEMM + log-sum-exp, 16 rows x (2 x 16 cols) per wave step.
// Per K-step: 1x A hi/lo fragment load feeds 2 column tiles -> 6 WMMAs.
// Lanes keep per-(row, col-class) running sums s += exp(x - RMAX); diagonal
// logit captured from the C fragment of whichever tile is the diagonal tile.
// 16-lane butterfly at the end; atomicAdd into Srow, plain store of Drow
// (each row's diagonal is owned by exactly one wave).
// ---------------------------------------------------------------------------
__global__ __launch_bounds__(256) void cs_gemm_lse(
    const _Float16* __restrict__ Ah, const _Float16* __restrict__ Al,
    const _Float16* __restrict__ Bh, const _Float16* __restrict__ Bl,
    float* __restrict__ Srow, float* __restrict__ Drow)
{
    const int lane    = threadIdx.x & 31;
    const int wv      = threadIdx.x >> 5;
    const int rowGrp  = blockIdx.x * 8 + wv;   // 0..511
    const int rowBase = rowGrp * 16;
    const int slice   = blockIdx.y;            // 0..NSLICES-1
    const int m       = lane & 15;             // row (A) / col (B) within tile
    const int h       = lane >> 4;             // half-wave selector

    // A fragment base: row = rowBase+m; 16 contiguous K at 16h (matches B)
    const _Float16* __restrict__ ahp = Ah + (size_t)(rowBase + m) * DIMK + 16 * h;
    const _Float16* __restrict__ alp = Al + (size_t)(rowBase + m) * DIMK + 16 * h;

    float s[8], d[8];
    #pragma unroll
    for (int j = 0; j < 8; ++j) { s[j] = 0.0f; d[j] = 0.0f; }

    const int diagTile = rowGrp;               // tile containing row==col
    const int ntBeg = slice * TILES_PER_SLICE;
    const int ntEnd = ntBeg + TILES_PER_SLICE;

    for (int nt0 = ntBeg; nt0 < ntEnd; nt0 += 2) {
        // B fragment: column nt*16+m of B^T == row of f2n; 16 contiguous K at 16h
        const _Float16* __restrict__ bhp = Bh + (size_t)(nt0 * 16 + m) * DIMK + 16 * h;
        const _Float16* __restrict__ blp = Bl + (size_t)(nt0 * 16 + m) * DIMK + 16 * h;

        v8f c0[2] = {};   // hi*hi accumulators (tile nt0, nt0+1)
        v8f cl[2] = {};   // (hi*lo' + lo'*hi) accumulators, scaled by LO_SCALE

        #pragma unroll 2
        for (int k = 0; k < DIMK; k += 32) {
            union { v16h v; v8h p[2]; } a_hi, a_lo;
            a_hi.p[0] = *(const v8h*)(ahp + k);
            a_hi.p[1] = *(const v8h*)(ahp + k + 8);
            a_lo.p[0] = *(const v8h*)(alp + k);
            a_lo.p[1] = *(const v8h*)(alp + k + 8);
            v16h bh0 = *(const v16h*)(bhp + k);
            v16h bl0 = *(const v16h*)(blp + k);
            v16h bh1 = *(const v16h*)(bhp + 16 * DIMK + k);
            v16h bl1 = *(const v16h*)(blp + 16 * DIMK + k);

            c0[0] = __builtin_amdgcn_wmma_f32_16x16x32_f16(
                        false, a_hi.v, false, bh0, (short)0, c0[0], false, false);
            c0[1] = __builtin_amdgcn_wmma_f32_16x16x32_f16(
                        false, a_hi.v, false, bh1, (short)0, c0[1], false, false);
            cl[0] = __builtin_amdgcn_wmma_f32_16x16x32_f16(
                        false, a_hi.v, false, bl0, (short)0, cl[0], false, false);
            cl[1] = __builtin_amdgcn_wmma_f32_16x16x32_f16(
                        false, a_hi.v, false, bl1, (short)0, cl[1], false, false);
            cl[0] = __builtin_amdgcn_wmma_f32_16x16x32_f16(
                        false, a_lo.v, false, bh0, (short)0, cl[0], false, false);
            cl[1] = __builtin_amdgcn_wmma_f32_16x16x32_f16(
                        false, a_lo.v, false, bh1, (short)0, cl[1], false, false);
        }

        #pragma unroll
        for (int t = 0; t < 2; ++t) {
            const int nt = nt0 + t;
            #pragma unroll
            for (int j = 0; j < 8; ++j) {
                // C layout: VGPR j holds row (j + 8h), col (lane&15)
                float x = (c0[t][j] + cl[t][j] * INV_LO_SCALE) * INV_TEMP;
                s[j] += __expf(x - RMAX);      // bounded: never over/underflows
                if (nt == diagTile) {
                    if (m == (j | (h << 3))) d[j] = x;   // row==col within tile
                }
            }
        }
    }

    // Butterfly sum across the 16-lane half (masks<16 never cross halves)
    #pragma unroll
    for (int j = 0; j < 8; ++j) {
        float sv = s[j], dv = d[j];
        #pragma unroll
        for (int mk = 1; mk <= 8; mk <<= 1) {
            sv += __shfl_xor(sv, mk, 32);
            dv += __shfl_xor(dv, mk, 32);
        }
        s[j] = sv; d[j] = dv;
    }

    if (m == 0) {   // lanes 0 (rows base..base+7) and 16 (rows base+8..base+15)
        const bool hasDiag = (diagTile >= ntBeg) && (diagTile < ntEnd);
        #pragma unroll
        for (int j = 0; j < 8; ++j) {
            const int row = rowBase + j + 8 * h;
            atomicAdd(&Srow[row], s[j]);
            if (hasDiag) Drow[row] = d[j];
        }
    }
}

// ---------------------------------------------------------------------------
// Kernel 3: loss = mean_r( RMAX + log(sum_r) - diag_r )
// ---------------------------------------------------------------------------
__global__ __launch_bounds__(256) void cs_finish(
    const float* __restrict__ Srow, const float* __restrict__ Drow,
    float* __restrict__ out)
{
    __shared__ float red[8];
    const int tid = threadIdx.x;
    float p = 0.0f;
    for (int r = tid; r < NROWS; r += 256)
        p += (RMAX + __logf(Srow[r])) - Drow[r];
    #pragma unroll
    for (int mk = 1; mk < 32; mk <<= 1) p += __shfl_xor(p, mk, 32);
    if ((tid & 31) == 0) red[tid >> 5] = p;
    __syncthreads();
    if (tid == 0) {
        float t = 0.0f;
        #pragma unroll
        for (int i = 0; i < 8; ++i) t += red[i];
        out[0] = t * (1.0f / (float)NROWS);
    }
}

// ---------------------------------------------------------------------------
extern "C" void kernel_launch(void* const* d_in, const int* in_sizes, int n_in,
                              void* d_out, int out_size, void* d_ws, size_t ws_size,
                              hipStream_t stream)
{
    (void)in_sizes; (void)n_in; (void)out_size; (void)ws_size;

    const float* f1 = (const float*)d_in[0];
    const float* f2 = (const float*)d_in[1];
    float* out = (float*)d_out;

    char* ws = (char*)d_ws;
    const size_t halfBytes = (size_t)NROWS * DIMK * sizeof(_Float16); // 12.58 MB
    _Float16* Ah = (_Float16*)(ws + 0 * halfBytes);
    _Float16* Al = (_Float16*)(ws + 1 * halfBytes);
    _Float16* Bh = (_Float16*)(ws + 2 * halfBytes);
    _Float16* Bl = (_Float16*)(ws + 3 * halfBytes);
    float* Srow  = (float*)(ws + 4 * halfBytes);
    float* Drow  = Srow + NROWS;
    // total workspace: ~50.4 MB

    cs_normalize_split<<<dim3(2 * NROWS), dim3(256), 0, stream>>>(
        f1, f2, Ah, Al, Bh, Bl, Srow, Drow);

    cs_gemm_lse<<<dim3(NTILES / 8, NSLICES), dim3(256), 0, stream>>>(
        Ah, Al, Bh, Bl, Srow, Drow);

    cs_finish<<<dim3(1), dim3(256), 0, stream>>>(Srow, Drow, out);
}
